// MatchSRNN_40785009443661
// MI455X (gfx1250) — matrix operand order repaired and checked by reading, not compiled, not run on hardware
//
#include <hip/hip_runtime.h>
#include <hip/hip_bf16.h>
#include <math.h>

// ---------------------------------------------------------------------------
// MatchSRNN on MI455X (gfx1250): bf16 WMMA for the two dominant einsums,
// f32 accumulation. Each wave now computes a 16x64 tile (4 accumulators) so
// one A-fragment load feeds 4 WMMAs -> fewer VMEM issues per matrix op.
// ---------------------------------------------------------------------------

typedef __bf16 bf16;
typedef __attribute__((ext_vector_type(16))) __bf16 v16bf;
typedef __attribute__((ext_vector_type(8)))  __bf16 v8bf;
typedef __attribute__((ext_vector_type(8)))  float  v8f;

#define BATCH 256
#define LEN   56
#define EMB   300
#define EP    320           // EMB padded to multiple of 32 (WMMA K)
#define DIMS  4
#define M1    (BATCH*LEN)   // 14336 rows of input1/input2
#define LPAD  64            // LEN padded to multiple of 16 for GEMM2 N
#define MT1   (M1/16)       // 896 M-tiles in GEMM1
#define NG1   (EP/64)       // 5 N-groups (4 subtiles of 16 each) in GEMM1
#define AT2   ((LEN*DIMS)/16) // 14 row-tiles in GEMM2 (224 rows exactly)
#define NTOT  (BATCH*LEN*LEN) // 802816 rows of the (N,4) view before reversal

// ---------------- conversion kernels (fp32 -> bf16, pad, transpose) --------

__global__ void k_cvt_x1(const float* __restrict__ in1, bf16* __restrict__ X1b) {
    int idx = blockIdx.x * blockDim.x + threadIdx.x;
    if (idx >= M1 * EP) return;
    int m = idx / EP, k = idx % EP;
    float v = (k < EMB) ? in1[m * EMB + k] : 0.0f;
    X1b[idx] = (bf16)v;
}

__global__ void k_cvt_x2(const float* __restrict__ in2, bf16* __restrict__ X2b) {
    int idx = blockIdx.x * blockDim.x + threadIdx.x;
    if (idx >= BATCH * LPAD * EP) return;
    int k = idx % EP;
    int r = (idx / EP) % LPAD;
    int b = idx / (EP * LPAD);
    float v = (r < LEN && k < EMB) ? in2[(b * LEN + r) * EMB + k] : 0.0f;
    X2b[idx] = (bf16)v;
}

// Tt[d][n][k] = T[d][k][n]  (B-matrix column-major so each lane's fragment
// is one contiguous 32B load), zero padded to 320x320.
__global__ void k_cvt_t(const float* __restrict__ T, bf16* __restrict__ Tt) {
    int idx = blockIdx.x * blockDim.x + threadIdx.x;
    if (idx >= DIMS * EP * EP) return;
    int k = idx % EP;
    int n = (idx / EP) % EP;
    int d = idx / (EP * EP);
    float v = (k < EMB && n < EMB) ? T[(d * EMB + k) * EMB + n] : 0.0f;
    Tt[idx] = (bf16)v;
}

// ---------------- small linear terms a(b,i,d), c(b,j,d) --------------------

__global__ void k_lin(const float* __restrict__ in1, const float* __restrict__ in2,
                      const float* __restrict__ Wlin,
                      float* __restrict__ a_arr, float* __restrict__ c_arr) {
    int tid = blockIdx.x * blockDim.x + threadIdx.x;
    if (tid >= 2 * M1 * DIMS) return;
    int sel = tid >= M1 * DIMS;
    int base = sel ? (tid - M1 * DIMS) : tid;
    int m = base / DIMS, d = base % DIMS;
    const float* x = sel ? (in2 + m * EMB) : (in1 + m * EMB);
    const float* w = Wlin + d * (2 * EMB) + (sel ? EMB : 0);
    float s = 0.0f;
    for (int e = 0; e < EMB; ++e) s = fmaf(x[e], w[e], s);
    if (sel) c_arr[base] = s; else a_arr[base] = s;
}

// ---------------- WMMA fragment helpers ------------------------------------
// A 16x32 bf16 fragment: lane holds row (lane&15), two contiguous 8-elem
// chunks at K offsets kg*16 and kg*16+8 (kg = lane>>4).

__device__ __forceinline__ v16bf load_a_frag(const bf16* __restrict__ rowPtr,
                                             int kb, int kg) {
    v8bf lo = *(const v8bf*)(rowPtr + kb + kg * 16);
    v8bf hi = *(const v8bf*)(rowPtr + kb + kg * 16 + 8);
    return __builtin_shufflevector(lo, hi, 0,1,2,3,4,5,6,7,8,9,10,11,12,13,14,15);
}

// ---------------- GEMM1: t1b[(m*4+d)][n] = X1b[m][:] . T[d][:][n] ----------
// One wave computes a 16x64 tile: 1 A fragment feeds 4 B subtiles per K chunk.

__global__ void __launch_bounds__(256)
k_gemm1(const bf16* __restrict__ X1b, const bf16* __restrict__ Tt,
        bf16* __restrict__ t1b) {
    int gtid = blockIdx.x * blockDim.x + threadIdx.x;
    int wid  = gtid >> 5;
    int lane = threadIdx.x & 31;
    int ng = wid % NG1;                  // N-group: 64 cols
    int mt = (wid / NG1) % MT1;
    int d  = wid / (NG1 * MT1);          // exact grid: d in [0,4)

    int l15 = lane & 15;
    int kg  = lane >> 4;
    const bf16* aRow = X1b + (mt * 16 + l15) * EP;
    const bf16* bBase = Tt + ((size_t)(d * EP + ng * 64 + l15)) * EP;

    v8f acc[4];
#pragma unroll
    for (int t = 0; t < 4; ++t) acc[t] = (v8f){0.f,0.f,0.f,0.f,0.f,0.f,0.f,0.f};

#pragma unroll
    for (int kb = 0; kb < EP; kb += 32) {
        v16bf afr = load_a_frag(aRow, kb, kg);
#pragma unroll
        for (int t = 0; t < 4; ++t) {
            v16bf bfr = *(const v16bf*)(bBase + (size_t)t * 16 * EP + kb + kg * 16);
            acc[t] = __builtin_amdgcn_wmma_f32_16x16x32_bf16(
                         false, afr, false, bfr, (short)0, acc[t], false, false);
        }
    }

    int m0 = mt * 16 + (lane >> 4) * 8;
#pragma unroll
    for (int t = 0; t < 4; ++t) {
        int col = ng * 64 + t * 16 + l15;
#pragma unroll
        for (int r = 0; r < 8; ++r) {
            int m = m0 + r;
            t1b[(size_t)(m * DIMS + d) * EP + col] = (bf16)acc[t][r];
        }
    }
}

// ---------------- GEMM2 + bilinear epilogue + reversal scatter -------------
// Per batch b: rows (i*4+d) of t1b (224 x 320) times X2b[b]^T (320 x 64pad).
// One wave computes a 16x64 tile = the whole padded j range for 16 (i,d) rows.

__global__ void __launch_bounds__(256)
k_gemm2(const bf16* __restrict__ t1b, const bf16* __restrict__ X2b,
        const float* __restrict__ a_arr, const float* __restrict__ c_arr,
        const float* __restrict__ b_lin, float* __restrict__ q) {
    int gtid = blockIdx.x * blockDim.x + threadIdx.x;
    int wid  = gtid >> 5;
    int lane = threadIdx.x & 31;
    int at = wid % AT2;
    int b  = wid / AT2;                  // exact grid: b in [0,256)

    int l15 = lane & 15;
    int kg  = lane >> 4;
    const bf16* aRow  = t1b + (size_t)(b * (LEN * DIMS) + at * 16 + l15) * EP;
    const bf16* bBase = X2b + ((size_t)(b * LPAD) + l15) * EP;

    v8f acc[4];
#pragma unroll
    for (int t = 0; t < 4; ++t) acc[t] = (v8f){0.f,0.f,0.f,0.f,0.f,0.f,0.f,0.f};

#pragma unroll
    for (int kb = 0; kb < EP; kb += 32) {
        v16bf afr = load_a_frag(aRow, kb, kg);
#pragma unroll
        for (int t = 0; t < 4; ++t) {
            v16bf bfr = *(const v16bf*)(bBase + (size_t)t * 16 * EP + kb + kg * 16);
            acc[t] = __builtin_amdgcn_wmma_f32_16x16x32_bf16(
                         false, afr, false, bfr, (short)0, acc[t], false, false);
        }
    }

    int ar0 = at * 16 + (lane >> 4) * 8;
#pragma unroll
    for (int t = 0; t < 4; ++t) {
        int j = t * 16 + l15;            // output column (padded to 64)
        if (j < LEN) {
#pragma unroll
            for (int r = 0; r < 8; ++r) {
                int ar = ar0 + r;        // = i*4 + d
                int i = ar >> 2, d = ar & 3;
                float v = acc[t][r]
                        + a_arr[(b * LEN + i) * DIMS + d]
                        + c_arr[(b * LEN + j) * DIMS + d]
                        + b_lin[d];
                v = fmaxf(v, 0.0f);      // ReLU before reshape
                // reference: reshape(-1,4)[::-1].reshape(B,4,56,56)
                int ro = b * (LEN * LEN) + i * LEN + j;
                int f  = (NTOT - 1 - ro) * 4 + d;  // flat idx into (B,4,56,56)
                q[f] = v;
            }
        }
    }
}

// ---------------- conv1 (4->16, 10x10, s2) + relu ; pool 2x2 ---------------

__global__ void k_conv1(const float* __restrict__ q, const float* __restrict__ W,
                        const float* __restrict__ bias, float* __restrict__ out) {
    int o = blockIdx.x * blockDim.x + threadIdx.x;
    if (o >= BATCH * 16 * 24 * 24) return;
    int ox = o % 24, oy = (o / 24) % 24, oc = (o / 576) % 16, b = o / (576 * 16);
    float s = bias[oc];
    const float* qb = q + (size_t)b * 4 * 56 * 56;
    const float* wc = W + (size_t)oc * 4 * 100;
    for (int ic = 0; ic < 4; ++ic) {
        const float* qp = qb + ic * 56 * 56 + (oy * 2) * 56 + (ox * 2);
        const float* wp = wc + ic * 100;
#pragma unroll
        for (int kh = 0; kh < 10; ++kh)
#pragma unroll
            for (int kw = 0; kw < 10; ++kw)
                s = fmaf(qp[kh * 56 + kw], wp[kh * 10 + kw], s);
    }
    out[o] = fmaxf(s, 0.0f);
}

__global__ void k_pool1(const float* __restrict__ x, float* __restrict__ y) {
    int p = blockIdx.x * blockDim.x + threadIdx.x;
    if (p >= BATCH * 16 * 12 * 12) return;
    int px = p % 12, py = (p / 12) % 12, c = (p / 144) % 16, b = p / (144 * 16);
    const float* xp = x + (((size_t)(b * 16 + c)) * 24 + py * 2) * 24 + px * 2;
    float m = fmaxf(fmaxf(xp[0], xp[1]), fmaxf(xp[24], xp[25]));
    y[p] = m;
}

// ---------------- conv2 (16->24, 5x5, s1) + relu ; pool 2x2 ----------------

__global__ void k_conv2(const float* __restrict__ p1, const float* __restrict__ W,
                        const float* __restrict__ bias, float* __restrict__ out) {
    int o = blockIdx.x * blockDim.x + threadIdx.x;
    if (o >= BATCH * 24 * 8 * 8) return;
    int ox = o % 8, oy = (o / 8) % 8, oc = (o / 64) % 24, b = o / (64 * 24);
    float s = bias[oc];
    const float* pb = p1 + (size_t)b * 16 * 144;
    const float* wc = W + (size_t)oc * 16 * 25;
    for (int ic = 0; ic < 16; ++ic) {
        const float* pp = pb + ic * 144 + oy * 12 + ox;
        const float* wp = wc + ic * 25;
#pragma unroll
        for (int kh = 0; kh < 5; ++kh)
#pragma unroll
            for (int kw = 0; kw < 5; ++kw)
                s = fmaf(pp[kh * 12 + kw], wp[kh * 5 + kw], s);
    }
    out[o] = fmaxf(s, 0.0f);
}

__global__ void k_pool2(const float* __restrict__ x, float* __restrict__ y) {
    int p = blockIdx.x * blockDim.x + threadIdx.x;
    if (p >= BATCH * 24 * 4 * 4) return;
    int px = p % 4, py = (p / 4) % 4, c = (p / 16) % 24, b = p / (16 * 24);
    const float* xp = x + (((size_t)(b * 24 + c)) * 8 + py * 2) * 8 + px * 2;
    float m = fmaxf(fmaxf(xp[0], xp[1]), fmaxf(xp[8], xp[9]));
    y[(size_t)b * 384 + c * 16 + py * 4 + px] = m;   // matches reshape(B,-1)
}

// ---------------- head: fc(384->10) tanh, fc(10->2) tanh, softmax ----------

__global__ void k_head(const float* __restrict__ p2,
                       const float* __restrict__ Wc1, const float* __restrict__ bc1,
                       const float* __restrict__ Wc2, const float* __restrict__ bc2,
                       float* __restrict__ out) {
    int b = blockIdx.x * blockDim.x + threadIdx.x;
    if (b >= BATCH) return;
    const float* x = p2 + (size_t)b * 384;
    float h[10];
    for (int u = 0; u < 10; ++u) {
        float s = bc1[u];
        const float* w = Wc1 + u * 384;
        for (int e = 0; e < 384; ++e) s = fmaf(x[e], w[e], s);
        h[u] = tanhf(s);
    }
    float z[2];
    for (int v = 0; v < 2; ++v) {
        float s = bc2[v];
        for (int u = 0; u < 10; ++u) s = fmaf(h[u], Wc2[v * 10 + u], s);
        z[v] = tanhf(s);
    }
    float m = fmaxf(z[0], z[1]);
    float e0 = expf(z[0] - m), e1 = expf(z[1] - m);
    float inv = 1.0f / (e0 + e1);
    out[b * 2 + 0] = e0 * inv;
    out[b * 2 + 1] = e1 * inv;
}

// ---------------------------------------------------------------------------

extern "C" void kernel_launch(void* const* d_in, const int* in_sizes, int n_in,
                              void* d_out, int out_size, void* d_ws, size_t ws_size,
                              hipStream_t stream) {
    const float* in1   = (const float*)d_in[0];   // (256,56,300)
    const float* in2   = (const float*)d_in[1];   // (256,56,300)
    const float* T     = (const float*)d_in[2];   // (4,300,300)
    const float* Wlin  = (const float*)d_in[3];   // (4,600)
    const float* blin  = (const float*)d_in[4];   // (4,)
    const float* W1    = (const float*)d_in[5];   // (16,4,10,10)
    const float* b1    = (const float*)d_in[6];
    const float* W2    = (const float*)d_in[7];   // (24,16,5,5)
    const float* b2    = (const float*)d_in[8];
    const float* Wc1   = (const float*)d_in[9];   // (10,384)
    const float* bc1   = (const float*)d_in[10];
    const float* Wc2   = (const float*)d_in[11];  // (2,10)
    const float* bc2   = (const float*)d_in[12];
    float* out = (float*)d_out;                   // (256,2)

    char* ws = (char*)d_ws;
    size_t off = 0;
    auto alloc = [&](size_t bytes) -> void* {
        void* p = ws + off;
        off += (bytes + 255) & ~(size_t)255;
        return p;
    };
    bf16*  X1b   = (bf16*)alloc((size_t)M1 * EP * sizeof(bf16));           // 9.2 MB
    bf16*  X2b   = (bf16*)alloc((size_t)BATCH * LPAD * EP * sizeof(bf16)); // 10.5 MB
    bf16*  Tt    = (bf16*)alloc((size_t)DIMS * EP * EP * sizeof(bf16));    // 0.8 MB
    bf16*  t1b   = (bf16*)alloc((size_t)M1 * DIMS * EP * sizeof(bf16));    // 36.7 MB
    float* a_arr = (float*)alloc((size_t)M1 * DIMS * sizeof(float));
    float* c_arr = (float*)alloc((size_t)M1 * DIMS * sizeof(float));
    float* q     = (float*)alloc((size_t)BATCH * 4 * 56 * 56 * sizeof(float));
    float* c1o   = (float*)alloc((size_t)BATCH * 16 * 24 * 24 * sizeof(float));
    float* p1    = (float*)alloc((size_t)BATCH * 16 * 12 * 12 * sizeof(float));
    float* c2o   = (float*)alloc((size_t)BATCH * 24 * 8 * 8 * sizeof(float));
    float* p2    = (float*)alloc((size_t)BATCH * 384 * sizeof(float));
    (void)ws_size; (void)in_sizes; (void)n_in; (void)out_size;

    const int TB = 256;
    k_cvt_x1<<<(M1 * EP + TB - 1) / TB, TB, 0, stream>>>(in1, X1b);
    k_cvt_x2<<<(BATCH * LPAD * EP + TB - 1) / TB, TB, 0, stream>>>(in2, X2b);
    k_cvt_t <<<(DIMS * EP * EP + TB - 1) / TB, TB, 0, stream>>>(T, Tt);
    k_lin   <<<(2 * M1 * DIMS + TB - 1) / TB, TB, 0, stream>>>(in1, in2, Wlin, a_arr, c_arr);

    // GEMM1: 896 * 5 * 4 = 17920 waves = 2240 blocks * 8 waves (exact)
    k_gemm1<<<(MT1 * NG1 * DIMS) / 8, TB, 0, stream>>>(X1b, Tt, t1b);
    // GEMM2: 256 * 14 = 3584 waves = 448 blocks * 8 waves (exact)
    k_gemm2<<<(BATCH * AT2) / 8, TB, 0, stream>>>(t1b, X2b, a_arr, c_arr, blin, q);

    k_conv1<<<(BATCH * 16 * 24 * 24 + TB - 1) / TB, TB, 0, stream>>>(q, W1, b1, c1o);
    k_pool1<<<(BATCH * 16 * 12 * 12 + TB - 1) / TB, TB, 0, stream>>>(c1o, p1);
    k_conv2<<<(BATCH * 24 * 8 * 8 + TB - 1) / TB, TB, 0, stream>>>(p1, W2, b2, c2o);
    k_pool2<<<(BATCH * 24 * 4 * 4 + TB - 1) / TB, TB, 0, stream>>>(c2o, p2);
    k_head <<<1, TB, 0, stream>>>(p2, Wc1, bc1, Wc2, bc2, out);
}
